// TSC_88699664597625
// MI455X (gfx1250) — matrix operand order repaired and checked by reading, not compiled, not run on hardware
//
#include <hip/hip_runtime.h>
#include <hip/hip_bf16.h>
#include <math.h>
#include <stdint.h>

// ---------------------------------------------------------------------------
// Problem constants (from the reference)
// ---------------------------------------------------------------------------
#define B_DIM     8
#define C_DIM     512
#define H_DIM     64
#define W_DIM     64
#define HW        (H_DIM * W_DIM)          // 4096
#define NPIX      (B_DIM * HW)             // 32768
#define K_DIM     3072
#define N_CLS     4
#define NQ        256
#define NSEL      (N_CLS * NQ)             // 1024 queries
#define LOGW      (K_DIM + 1)              // 3073 logits columns
#define INV_TEMP  (1.0f / 0.07f)

// WMMA tiling
#define KT_CNT    (C_DIM / 32)             // 16 K-tiles of 32
#define NT_CNT    (K_DIM / 16)             // 192 N-tiles of 16

typedef __attribute__((ext_vector_type(16))) __bf16 v16bf;
typedef __attribute__((ext_vector_type(8)))  float  v8f;

struct Frag32 { uint4 lo; uint4 hi; };     // 32 bytes == v16bf

// ---------------------------------------------------------------------------
// 1) Deterministic stand-in for the JAX categorical sampling + shuffle.
//    (threefry is not reproducible here; structure/data-movement is identical)
// ---------------------------------------------------------------------------
__global__ void TSC_sample_kernel(const long long* __restrict__ target,
                                  int* __restrict__ idx,
                                  int* __restrict__ labels) {
    int j = blockIdx.x * blockDim.x + threadIdx.x;
    if (j >= NSEL) return;
    // bijective "shuffle": odd multiplier mod 1024
    unsigned s = (unsigned)((j * 741 + 13) & (NSEL - 1));
    int cls = (int)(s >> 8);
    int q   = (int)(s & (NQ - 1));
    unsigned h = 0x9E3779B9u * (unsigned)(cls + 1) ^
                 0x85EBCA6Bu * (unsigned)(q + 17) ^ 0xC2B2AE35u;
    int p = 0;
    #pragma unroll 1
    for (int a = 0; a < 64; ++a) {          // rejection sample: P(hit) ~ 1/4
        h = h * 1664525u + 1013904223u;
        p = (int)((h >> 9) & (NPIX - 1));
        if ((int)target[p] == cls) break;
    }
    idx[j]    = p;
    labels[j] = cls;
}

// ---------------------------------------------------------------------------
// 2) Pack queue (f32 [C,K]) into WMMA B-fragment order, bf16.
//    Tile t = nt*KT_CNT + kt holds a 32x16 (KxN) tile as [lane 0..31][16 halfs]
//    lane<16: K = k0+0..7,16..23 (pairs per VGPR), lane>=16: K = k0+8..15,24..31
// ---------------------------------------------------------------------------
__global__ void TSC_prep_queue_kernel(const float* __restrict__ queue,
                                      __bf16* __restrict__ Bpack) {
    const int t  = blockIdx.x;              // 0 .. NT_CNT*KT_CNT-1
    const int nt = t >> 4;                  // KT_CNT == 16
    const int kt = t & 15;
    const int k0 = kt * 32;
    const int n0 = nt * 16;
    #pragma unroll
    for (int rep = 0; rep < 2; ++rep) {
        int ee   = threadIdx.x + rep * 256; // 0..511
        int lane = ee >> 4;
        int jj   = ee & 15;
        int c    = k0 + ((lane >= 16) ? 8 : 0) + ((jj < 8) ? jj : jj + 8);
        int col  = n0 + (lane & 15);
        float v  = queue[(size_t)c * K_DIM + col];
        Bpack[(size_t)t * 512 + ee] = (__bf16)v;
    }
}

// ---------------------------------------------------------------------------
// 3) Gather 1024 random pixels, L2-normalize (s and t), emit:
//    - Apack  : normalized feats_s, bf16 row-major [NSEL, C]
//    - logits[n*LOGW + 0] = l_pos / T
//    - outLabels[n]
// ---------------------------------------------------------------------------
__global__ void TSC_gather_kernel(const float* __restrict__ fs_raw,
                                  const float* __restrict__ ft_raw,
                                  const int*   __restrict__ idx,
                                  const int*   __restrict__ labels,
                                  __bf16* __restrict__ Apack,
                                  float*  __restrict__ logits,
                                  float*  __restrict__ outLabels) {
    __shared__ float rs[256];
    __shared__ float rt[256];
    __shared__ float rd[256];
    __shared__ float bc[2];

    const int n   = blockIdx.x;
    const int tid = threadIdx.x;
    const int p   = idx[n];
    const int b   = p >> 12;                // / HW
    const int hw  = p & (HW - 1);

    const size_t base = ((size_t)b * C_DIM) * HW + hw;
    const float fs0 = fs_raw[base + (size_t)tid * HW];
    const float fs1 = fs_raw[base + (size_t)(tid + 256) * HW];
    const float ft0 = ft_raw[base + (size_t)tid * HW];
    const float ft1 = ft_raw[base + (size_t)(tid + 256) * HW];

    rs[tid] = fs0 * fs0 + fs1 * fs1;
    rt[tid] = ft0 * ft0 + ft1 * ft1;
    rd[tid] = fs0 * ft0 + fs1 * ft1;
    __syncthreads();
    #pragma unroll
    for (int s = 128; s > 0; s >>= 1) {
        if (tid < s) {
            rs[tid] += rs[tid + s];
            rt[tid] += rt[tid + s];
            rd[tid] += rd[tid + s];
        }
        __syncthreads();
    }
    if (tid == 0) {
        float ns = fmaxf(sqrtf(rs[0]), 1e-12f);
        float nt = fmaxf(sqrtf(rt[0]), 1e-12f);
        bc[0] = 1.0f / ns;
        logits[(size_t)n * LOGW] = (rd[0] / (ns * nt)) * INV_TEMP; // l_pos
        outLabels[n] = (float)labels[n];
    }
    __syncthreads();
    const float inv_ns = bc[0];
    Apack[(size_t)n * C_DIM + tid]       = (__bf16)(fs0 * inv_ns);
    Apack[(size_t)n * C_DIM + tid + 256] = (__bf16)(fs1 * inv_ns);
}

// ---------------------------------------------------------------------------
// 4) mask[n,k] = (labels[n] == queue_labels[k])
// ---------------------------------------------------------------------------
__global__ void TSC_mask_kernel(const int* __restrict__ labels,
                                const long long* __restrict__ qlab,
                                float* __restrict__ mask) {
    const int n = blockIdx.y;
    const int k = blockIdx.x * blockDim.x + threadIdx.x;
    const int l = labels[n];
    mask[(size_t)n * K_DIM + k] = (l == (int)qlab[k]) ? 1.0f : 0.0f;
}

// ---------------------------------------------------------------------------
// 5) l_neg GEMM: logits[:,1:] = (A[1024,512] @ queue[512,3072]) / T
//    Block = 128 threads (4 wave32). Block owns 16 rows x 256 cols.
//    A slab (16x512 bf16 = 16KB) -> LDS via GLOBAL_LOAD_ASYNC_TO_LDS_B128
//    (ASYNCcnt path, no VGPR round-trip). K-loop software-pipelined: next
//    iteration's A/B fragments are loaded before the current WMMA group so
//    the s_wait lands after compute, not before each v_wmma.
// ---------------------------------------------------------------------------
__device__ inline v16bf TSC_load_a_frag(const __bf16* As, int lane, int kt) {
    const int row = lane & 15;
    const int off = (lane >> 4) << 3;                      // 0 or 8 halfs
    const __bf16* p = As + row * C_DIM + kt * 32 + off;
    Frag32 f;
    f.lo = *(const uint4*)(p);                              // K pairs 0..7
    f.hi = *(const uint4*)(p + 16);                         // K pairs 16..23
    return __builtin_bit_cast(v16bf, f);
}

__global__ void TSC_gemm_kernel(const __bf16* __restrict__ A,
                                const __bf16* __restrict__ Bpack,
                                float* __restrict__ logits) {
    __shared__ __align__(32) __bf16 As[16 * C_DIM];        // 16 KB

    const int mtile = blockIdx.x;                           // 0..63
    const int nblk  = blockIdx.y;                           // 0..11
    const int tid   = threadIdx.x;
    const int wave  = tid >> 5;
    const int lane  = tid & 31;

    // Stage the 16-row A slab straight into LDS (async path, ASYNCcnt).
    {
        const __bf16* src = A + (size_t)mtile * 16 * C_DIM;
        #pragma unroll
        for (int i = 0; i < 8; ++i) {
            unsigned           lds = (unsigned)(uintptr_t)As + (unsigned)((tid + i * 128) * 16);
            unsigned long long ga  = (unsigned long long)(uintptr_t)src +
                                     (unsigned long long)((tid + i * 128) * 16);
            asm volatile("global_load_async_to_lds_b128 %0, %1, off"
                         :: "v"(lds), "v"(ga) : "memory");
        }
        asm volatile("s_wait_asynccnt 0x0" ::: "memory");
    }
    __syncthreads();

    const int nt0 = nblk * 16 + wave * 4;                   // first of 4 N-tiles
    // Per-wave B base; tile t at +t*16KB, K-step at +1KB (folds to immediates).
    const __bf16* Bw = Bpack + ((size_t)nt0 * KT_CNT) * 512 + lane * 16;

    v8f acc[4] = {};

    // ---- software-pipelined K loop --------------------------------------
    v16bf a = TSC_load_a_frag(As, lane, 0);
    v16bf b[4];
    #pragma unroll
    for (int t = 0; t < 4; ++t)
        b[t] = *(const v16bf*)(Bw + (size_t)t * (KT_CNT * 512));

    #pragma unroll 2
    for (int kt = 0; kt < KT_CNT; ++kt) {
        const int ktn = (kt + 1 < KT_CNT) ? kt + 1 : kt;    // last iter: re-hit L0
        // issue next-iteration loads first (overlap with WMMA group below)
        v16bf an = TSC_load_a_frag(As, lane, ktn);
        v16bf bn[4];
        #pragma unroll
        for (int t = 0; t < 4; ++t)
            bn[t] = *(const v16bf*)(Bw + (size_t)t * (KT_CNT * 512) + (size_t)ktn * 512);
        if (kt + 2 < KT_CNT)                                // L2-distance prefetch
            __builtin_prefetch(Bw + (size_t)(kt + 2) * 512, 0, 1);

        #pragma unroll
        for (int t = 0; t < 4; ++t)
            acc[t] = __builtin_amdgcn_wmma_f32_16x16x32_bf16(
                false, a, false, b[t], (short)0, acc[t], false, false);

        a = an;
        #pragma unroll
        for (int t = 0; t < 4; ++t) b[t] = bn[t];
    }

    // Scatter C/D: VGPR r -> M = r (+8 for lanes 16..31), N = lane&15.
    const int row0 = mtile * 16 + ((lane >> 4) << 3);
    const int colb = 1 + (lane & 15);                       // col 0 is l_pos
    #pragma unroll
    for (int t = 0; t < 4; ++t) {
        const int col = colb + (nt0 + t) * 16;
        #pragma unroll
        for (int r = 0; r < 8; ++r)
            logits[(size_t)(row0 + r) * LOGW + col] = acc[t][r] * INV_TEMP;
    }
}

// ---------------------------------------------------------------------------
// Launch
// ---------------------------------------------------------------------------
extern "C" void kernel_launch(void* const* d_in, const int* in_sizes, int n_in,
                              void* d_out, int out_size, void* d_ws, size_t ws_size,
                              hipStream_t stream) {
    // inputs (setup_inputs order): pred_s, feat_s_raw, feat_t_raw, target,
    //                              queue, queue_labels
    const float*     fs_raw = (const float*)d_in[1];
    const float*     ft_raw = (const float*)d_in[2];
    const long long* target = (const long long*)d_in[3];
    const float*     queue  = (const float*)d_in[4];
    const long long* qlab   = (const long long*)d_in[5];

    float* logits    = (float*)d_out;                       // [1024, 3073]
    float* mask      = logits + (size_t)NSEL * LOGW;        // [1024, 3072]
    float* outLabels = mask + (size_t)NSEL * K_DIM;         // [1024]

    // workspace layout
    int*    ws_idx = (int*)d_ws;                                     //  4 KB
    int*    ws_lab = ws_idx + NSEL;                                  //  4 KB
    __bf16* Apack  = (__bf16*)((char*)d_ws + 8192);                  //  1 MB
    __bf16* Bpack  = (__bf16*)((char*)d_ws + 8192 +
                               (size_t)NSEL * C_DIM * 2);            //  3 MB

    TSC_sample_kernel<<<NSEL / 256, 256, 0, stream>>>(target, ws_idx, ws_lab);
    TSC_prep_queue_kernel<<<NT_CNT * KT_CNT, 256, 0, stream>>>(queue, Bpack);
    TSC_gather_kernel<<<NSEL, 256, 0, stream>>>(fs_raw, ft_raw, ws_idx, ws_lab,
                                                Apack, logits, outLabels);
    TSC_mask_kernel<<<dim3(K_DIM / 256, NSEL), 256, 0, stream>>>(ws_lab, qlab, mask);
    TSC_gemm_kernel<<<dim3(NSEL / 16, K_DIM / 256), 128, 0, stream>>>(Apack, Bpack,
                                                                      logits);
}
